// CausalAttentionBlock_49606872268962
// MI455X (gfx1250) — matrix operand order repaired and checked by reading, not compiled, not run on hardware
//
#include <hip/hip_runtime.h>
#include <hip/hip_bf16.h>

// ---------------------------------------------------------------------------
// Causal MHA block for MI455X (gfx1250), bf16 WMMA pipeline.
//   B=128 T=200 C=512 H=16 Dqk=64, v head dim = C.
// d_out = [ y (B*T*C f32) | att (B*H*T*T f32) ]
// Scratch carve (~943 MB): bf16 copies of x/W*, then q,k,v,y in bf16.
// CDNA5 specifics: v_wmma_f32_16x16x32_bf16, global_load_async_to_lds_b128
// (+ s_wait_asynccnt), global_prefetch_b8, A/B frags in contiguous k = 16*(lane/16)+e, C layout per ISA
// 7.12.2 (B tiles kept transposed in LDS so B-fragment reads are contiguous).
// ---------------------------------------------------------------------------

#define BB   128
#define TT   200
#define CC   512
#define HH   16
#define DK   64
#define MR   (BB*TT)          // 25600 rows
#define NQK  (HH*DK)          // 1024
#define NV   (HH*CC)          // 8192
#define TP   208              // T padded to 13 tiles of 16
#define TPK  224              // T padded to 7 k-steps of 32

typedef __bf16 bf16_t;
typedef __bf16 v16bf __attribute__((ext_vector_type(16)));
typedef float  v8f   __attribute__((ext_vector_type(8)));

#define WMMA_BF16(a, b, c) \
  __builtin_amdgcn_wmma_f32_16x16x32_bf16(false, (a), false, (b), (short)0, (c), false, false)

// A fragment (16x32 bf16, MxK) from row-major [row*ld + k] storage.
// lane(<16)=row M, holds K {kh*16..+15} -> one contiguous 32B run per
// lane (matches load_bT_frag k order).
__device__ __forceinline__ v16bf load_a_frag(const bf16_t* base, int ld, int lane) {
  int row = lane & 15, kh = (lane >> 4) & 1;
  const bf16_t* p = base + row * ld;
  v16bf a;
#pragma unroll
  for (int e = 0; e < 16; ++e) {
    int pg = e >> 1, sl = e & 1;
    int k  = kh * 16 + 2 * pg + sl;
    a[e] = p[k];
  }
  return a;
}

// B fragment (32x16, KxN) read from *transposed* storage: element
// B[k][n] = base[n*ld + k].  Lane holds col n fixed, K = kh*16+e -> one
// contiguous 32B per-lane read.
__device__ __forceinline__ v16bf load_bT_frag(const bf16_t* base, int ld, int lane) {
  int col = lane & 15, kh = (lane >> 4) & 1;
  const bf16_t* p = base + col * ld + kh * 16;
  v16bf b;
#pragma unroll
  for (int e = 0; e < 16; ++e) b[e] = p[e];
  return b;
}

// ---------------------------------------------------------------------------
__global__ void cvt_f32_to_bf16(const float* __restrict__ s, bf16_t* __restrict__ d, int n) {
  int i = blockIdx.x * blockDim.x + threadIdx.x;
  int st = gridDim.x * blockDim.x;
  for (; i < n; i += st) d[i] = (bf16_t)s[i];
}

// ---------------------------------------------------------------------------
// GEMM: out = A(MxK bf16) @ B(KxN bf16) + bias.  256 threads (8 waves),
// block tile 64(M) x 128(N); wave (mi,ni) computes a 32x32 tile via 2x2 WMMA.
// A staged row-major, B staged transposed (so both fragment reads are
// contiguous).  All M,N,K multiples of 64/128/32 at every call site.
// ---------------------------------------------------------------------------
__global__ void gemm_bf16(const bf16_t* __restrict__ A, const bf16_t* __restrict__ B,
                          const float* __restrict__ bias, int K, int lda, int ldb, int ldo,
                          bf16_t* __restrict__ outb, float* __restrict__ outf) {
  __shared__ bf16_t As[64 * 32];       // 4 KB, row-major [m][k]
  __shared__ bf16_t BsT[128 * 32];     // 8 KB, transposed [n][k]
  const int tid  = threadIdx.x;
  const int lane = tid & 31;
  const int wave = tid >> 5;
  const int mi = wave >> 2;            // 0..1
  const int ni = wave & 3;             // 0..3
  const int n0 = blockIdx.x * 128;
  const int m0 = blockIdx.y * 64;

  // staging roles
  const int ar = tid >> 2;             // A row 0..63
  const int ac = (tid & 3) * 8;        // A col chunk (16B)
  const int br = tid >> 3;             // B row (k) 0..31
  const int bc = (tid & 7) * 16;       // B col chunk (32B)

  v8f acc00 = {}, acc01 = {}, acc10 = {}, acc11 = {};

  for (int k0 = 0; k0 < K; k0 += 32) {
    const bf16_t* apg = A + (size_t)(m0 + ar) * lda + k0 + ac;
    const bf16_t* bpg = B + (size_t)(k0 + br) * ldb + n0 + bc;
    if (k0 + 32 < K) {                 // prefetch next K panel
      __builtin_prefetch(apg + 32, 0, 1);
      __builtin_prefetch(bpg + (size_t)32 * ldb, 0, 1);
    }
    // A: 64x32, each thread one 16B chunk
    *(uint4*)(As + ar * 32 + ac) = *(const uint4*)apg;
    // B: 32x128 -> BsT[128][32], each thread 16 elements transposed
    union { uint4 u[2]; bf16_t h[16]; } tb;
    tb.u[0] = ((const uint4*)bpg)[0];
    tb.u[1] = ((const uint4*)bpg)[1];
#pragma unroll
    for (int i = 0; i < 16; ++i) BsT[(bc + i) * 32 + br] = tb.h[i];
    __syncthreads();

    const bf16_t* ab = As + (mi * 32) * 32;
    const bf16_t* bb = BsT + (ni * 32) * 32;
    v16bf a0 = load_a_frag(ab,           32, lane);
    v16bf a1 = load_a_frag(ab + 16 * 32, 32, lane);
    v16bf b0 = load_bT_frag(bb,           32, lane);
    v16bf b1 = load_bT_frag(bb + 16 * 32, 32, lane);
    acc00 = WMMA_BF16(a0, b0, acc00);
    acc01 = WMMA_BF16(a0, b1, acc01);
    acc10 = WMMA_BF16(a1, b0, acc10);
    acc11 = WMMA_BF16(a1, b1, acc11);
    __syncthreads();
  }

  const int col = lane & 15, kh = (lane >> 4) & 1;
  const int mw = m0 + mi * 32;
  const int nw = n0 + ni * 32;
  const float bv0 = bias[nw + col];
  const float bv1 = bias[nw + 16 + col];
#pragma unroll
  for (int r = 0; r < 8; ++r) {
    int m = mw + r + 8 * kh;
    size_t ro0 = (size_t)m * ldo, ro1 = (size_t)(m + 16) * ldo;
    float v00 = acc00[r] + bv0, v01 = acc01[r] + bv1;
    float v10 = acc10[r] + bv0, v11 = acc11[r] + bv1;
    if (outf) {
      outf[ro0 + nw + col] = v00;       outf[ro0 + nw + 16 + col] = v01;
      outf[ro1 + nw + col] = v10;       outf[ro1 + nw + 16 + col] = v11;
    } else {
      outb[ro0 + nw + col] = (bf16_t)v00;  outb[ro0 + nw + 16 + col] = (bf16_t)v01;
      outb[ro1 + nw + col] = (bf16_t)v10;  outb[ro1 + nw + 16 + col] = (bf16_t)v11;
    }
  }
}

// ---------------------------------------------------------------------------
// Fused attention: grid (13 t-tiles, H, B), 256 threads (8 waves).
//   k_h panel staged with global_load_async_to_lds_b128 (ASYNCcnt),
//   scores (WMMA q@k^T) -> fp32 masked softmax -> att out + y strip (WMMA).
// Static LDS = 58.4 KB.
// ---------------------------------------------------------------------------
__global__ void attn_kernel(const bf16_t* __restrict__ qg, const bf16_t* __restrict__ kg,
                            const bf16_t* __restrict__ vg, float* __restrict__ att_g,
                            bf16_t* __restrict__ yg) {
  __shared__ bf16_t q_s[16 * DK];        //  2 KB  q tile (rows padded w/ 0)
  __shared__ bf16_t k_s[TP * DK];        // 26 KB  full k_h (rows 200..207 = 0)
  __shared__ float  sc[16 * TP];         // 13 KB  raw scaled scores
  __shared__ bf16_t att_s[16 * TPK];     //  7 KB  softmaxed probs (bf16)
  __shared__ bf16_t vbT[128 * 32];       //  8 KB  staged v slab, transposed
  __shared__ float  red[16 * 17];        //  1 KB  softmax reductions

  const int tid   = threadIdx.x;
  const int lane  = tid & 31;
  const int wave  = tid >> 5;
  const int ttile = blockIdx.x;          // 0..12
  const int h     = blockIdx.y;
  const int b     = blockIdx.z;

  // ---- Phase 1: stage q tile (VALU path) + k_h panel (async DMA to LDS) ----
  {
    // k_h: rows 0..199 are 1600 16B chunks, row-contiguous in LDS (128B rows)
    const bf16_t* kbase = kg + (size_t)b * TT * NQK + (size_t)h * DK;
    unsigned lds0 = (unsigned)(unsigned long long)&k_s[0];
    for (int c = tid; c < 1600; c += 256) {
      int s = c >> 3;
      unsigned ldsa = lds0 + (unsigned)(c * 16);
      unsigned long long ga =
          (unsigned long long)(kbase + (size_t)s * NQK) + (unsigned)((c & 7) * 16);
      asm volatile("global_load_async_to_lds_b128 %0, %1, off"
                   :: "v"(ldsa), "v"(ga) : "memory");
    }
    for (int i = tid; i < 8 * DK; i += 256)          // zero pad rows 200..207
      k_s[200 * DK + i] = (bf16_t)0.0f;
    for (int i = tid; i < 16 * DK; i += 256) {
      int r = i >> 6, c = i & 63, t = ttile * 16 + r;
      q_s[i] = (t < TT) ? qg[(size_t)(b * TT + t) * NQK + h * DK + c] : (bf16_t)0.0f;
    }
    asm volatile("s_wait_asynccnt 0x0" ::: "memory");
  }
  __syncthreads();

  // ---- Phase 2: scores = (q @ k^T) * 1/sqrt(64) ----
  for (int st = wave; st < 13; st += 8) {
    v8f acc = {};
#pragma unroll
    for (int kk = 0; kk < DK; kk += 32) {
      v16bf a   = load_a_frag(q_s + kk, DK, lane);
      v16bf bfr = load_bT_frag(k_s + (size_t)(st * 16) * DK + kk, DK, lane);
      acc = WMMA_BF16(a, bfr, acc);
    }
    int col = lane & 15, kh = (lane >> 4) & 1;
#pragma unroll
    for (int r = 0; r < 8; ++r) {
      int m = r + 8 * kh;
      sc[m * TP + st * 16 + col] = acc[r] * 0.125f;
    }
  }
  __syncthreads();

  // ---- Phase 3: causal softmax (fp32); emit att (f32 global) + att_s (bf16)
  {
    const int r = tid >> 4, j = tid & 15;
    const int t = ttile * 16 + r;
    const bool valid = (t < TT);
    float mx = -3.0e38f;
    if (valid) for (int s = j; s <= t; s += 16) mx = fmaxf(mx, sc[r * TP + s]);
    red[r * 17 + j] = mx;
    __syncthreads();
    if (j == 0) {
      float mm = -3.0e38f;
      for (int q = 0; q < 16; ++q) mm = fmaxf(mm, red[r * 17 + q]);
      red[r * 17 + 16] = mm;
    }
    __syncthreads();
    const float rowmax = red[r * 17 + 16];
    __syncthreads();
    float sm = 0.0f;
    if (valid) for (int s = j; s <= t; s += 16) sm += __expf(sc[r * TP + s] - rowmax);
    red[r * 17 + j] = sm;
    __syncthreads();
    if (j == 0) {
      float tot = 0.0f;
      for (int q = 0; q < 16; ++q) tot += red[r * 17 + q];
      red[r * 17 + 16] = tot;
    }
    __syncthreads();
    const float inv = valid ? 1.0f / red[r * 17 + 16] : 0.0f;
    for (int s = j; s < TPK; s += 16) {
      float av = 0.0f;
      if (valid && s <= t) av = __expf(sc[r * TP + s] - rowmax) * inv;
      att_s[r * TPK + s] = (bf16_t)av;
      if (valid && s < TT)
        att_g[((size_t)((b * HH + h) * TT + t)) * TT + s] = av;   // 0 above diag
    }
  }
  __syncthreads();

  // ---- Phase 4: y strip = att(16xTPK) @ v_h(TPKxCC), 4 column passes of 128
  for (int npass = 0; npass < 4; ++npass) {
    const int n0 = npass * 128;
    v8f acc = {};
    for (int kstep = 0; kstep < TPK / 32; ++kstep) {
      // stage v rows [kstep*32,+32) x cols [n0,+128) transposed into vbT
      int idx0 = tid * 16;
      int vr = idx0 >> 7;              // v row within slab, 0..31
      int vc = idx0 & 127;             // col chunk base
      int s = kstep * 32 + vr;
      if (s < TT) {
        union { uint4 u[2]; bf16_t h[16]; } tv;
        const uint4* src = (const uint4*)(vg + (size_t)(b * TT + s) * NV + h * CC + n0 + vc);
        tv.u[0] = src[0]; tv.u[1] = src[1];
#pragma unroll
        for (int i = 0; i < 16; ++i) vbT[(vc + i) * 32 + vr] = tv.h[i];
      } else {
#pragma unroll
        for (int i = 0; i < 16; ++i) vbT[(vc + i) * 32 + vr] = (bf16_t)0.0f;
      }
      __syncthreads();
      v16bf a   = load_a_frag(att_s + kstep * 32, TPK, lane);
      v16bf bfr = load_bT_frag(vbT + (wave * 16) * 32, 32, lane);
      acc = WMMA_BF16(a, bfr, acc);
      __syncthreads();
    }
    const int col = n0 + wave * 16 + (lane & 15);
    const int kh = (lane >> 4) & 1;
#pragma unroll
    for (int r = 0; r < 8; ++r) {
      int m = r + 8 * kh, t = ttile * 16 + m;
      if (t < TT) yg[(size_t)(b * TT + t) * NV + h * CC + col] = (bf16_t)acc[r];
    }
  }
}

// ---------------------------------------------------------------------------
extern "C" void kernel_launch(void* const* d_in, const int* in_sizes, int n_in,
                              void* d_out, int out_size, void* d_ws, size_t ws_size,
                              hipStream_t stream) {
  const float* x  = (const float*)d_in[0];
  const float* Wq = (const float*)d_in[1];
  const float* bq = (const float*)d_in[2];
  const float* Wk = (const float*)d_in[3];
  const float* bk = (const float*)d_in[4];
  const float* Wv = (const float*)d_in[5];
  const float* bv = (const float*)d_in[6];
  const float* Wp = (const float*)d_in[7];
  const float* bp = (const float*)d_in[8];

  float* out_y   = (float*)d_out;
  float* out_att = out_y + (size_t)BB * TT * CC;

  // scratch carve (bf16 regions), ~943 MB total
  char* ws = (char*)d_ws;
  size_t o = 0;
  bf16_t* xb  = (bf16_t*)(ws + o); o += (size_t)MR * CC * 2;      // x  bf16
  bf16_t* wqb = (bf16_t*)(ws + o); o += (size_t)CC * NQK * 2;     // Wq bf16
  bf16_t* wkb = (bf16_t*)(ws + o); o += (size_t)CC * NQK * 2;     // Wk bf16
  bf16_t* wvb = (bf16_t*)(ws + o); o += (size_t)CC * NV * 2;      // Wv bf16
  bf16_t* wpb = (bf16_t*)(ws + o); o += (size_t)NV * CC * 2;      // Wp bf16
  bf16_t* qb  = (bf16_t*)(ws + o); o += (size_t)MR * NQK * 2;     // q  bf16
  bf16_t* kb  = (bf16_t*)(ws + o); o += (size_t)MR * NQK * 2;     // k  bf16
  bf16_t* vb  = (bf16_t*)(ws + o); o += (size_t)MR * NV * 2;      // v  bf16
  bf16_t* yb  = (bf16_t*)(ws + o); o += (size_t)MR * NV * 2;      // y  bf16

  // 1) fp32 -> bf16 conversions
  cvt_f32_to_bf16<<<2048, 256, 0, stream>>>(x,  xb,  MR * CC);
  cvt_f32_to_bf16<<<1024, 256, 0, stream>>>(Wq, wqb, CC * NQK);
  cvt_f32_to_bf16<<<1024, 256, 0, stream>>>(Wk, wkb, CC * NQK);
  cvt_f32_to_bf16<<<2048, 256, 0, stream>>>(Wv, wvb, CC * NV);
  cvt_f32_to_bf16<<<2048, 256, 0, stream>>>(Wp, wpb, NV * CC);

  // 2) projections (bf16 WMMA, fp32 accum); block tile 64x128
  gemm_bf16<<<dim3(NQK / 128, MR / 64), 256, 0, stream>>>(xb, wqb, bq, CC, CC, NQK, NQK, qb, nullptr);
  gemm_bf16<<<dim3(NQK / 128, MR / 64), 256, 0, stream>>>(xb, wkb, bk, CC, CC, NQK, NQK, kb, nullptr);
  gemm_bf16<<<dim3(NV  / 128, MR / 64), 256, 0, stream>>>(xb, wvb, bv, CC, CC, NV,  NV,  vb, nullptr);

  // 3) fused scores -> softmax -> att out -> att @ v
  attn_kernel<<<dim3(13, HH, BB), 256, 0, stream>>>(qb, kb, vb, out_att, yb);

  // 4) output projection -> f32 directly into d_out
  gemm_bf16<<<dim3(CC / 128, MR / 64), 256, 0, stream>>>(yb, wpb, bp, NV, NV, CC, CC, nullptr, out_y);
}
